// IDR_DCRB_16655883174451
// MI455X (gfx1250) — matrix operand order repaired and checked by reading, not compiled, not run on hardware
//
#include <hip/hip_runtime.h>

// ---------------------------------------------------------------------------
// Fused IDR_DCRB for MI455X (gfx1250, wave32, WMMA).
//   k1 : hid = leaky(d@W1^T); dk = hid@W2^T           (tiny)
//   k1b: repack conv_w f32 [n][ci][3][3] -> bf16 Wp[t][n][ci]
//   k2 : depthwise dynamic 3x3 conv + leaky -> tmp (bf16, halves HBM traffic)
//   k3 : dense 3x3 conv as implicit GEMM on v_wmma_f32_16x16x32_bf16,
//        3 input rows staged to LDS ONCE, 9 taps x 2 K-chunks fully unrolled
//        (18 WMMAs back-to-back), fused bias + residual-add epilogue.
// ---------------------------------------------------------------------------

typedef __attribute__((ext_vector_type(16))) __bf16 v16bf;
typedef __attribute__((ext_vector_type(8)))  __bf16 v8bf;
typedef __attribute__((ext_vector_type(8)))  float  v8f;
typedef __attribute__((ext_vector_type(4)))  float  v4f;

#define BB 16
#define CC 64
#define HH 128
#define WW 128

// workspace layout (bytes):
//   [0,      36864)  dk   f32   [B*C][9]
//   [40960, 114688)  Wp   bf16  [9][64 out][64 in]
//   [131072, +32MB)  tmp  bf16  [B][C][H][W]
#define WS_WP_OFF  40960
#define WS_TMP_OFF 131072

__device__ __forceinline__ float leakyf(float v) { return v >= 0.f ? v : 0.1f * v; }

// ---- k1: dynamic depthwise kernels ----------------------------------------
__global__ __launch_bounds__(64) void prep_dyn(const float* __restrict__ d,
                                               const float* __restrict__ W1,
                                               const float* __restrict__ W2,
                                               float* __restrict__ dk) {
    __shared__ float hid[CC];
    int b = blockIdx.x, c = threadIdx.x;
    float a = 0.f;
    for (int j = 0; j < CC; ++j) a += d[b * CC + j] * W1[c * CC + j];
    hid[c] = leakyf(a);
    __syncthreads();
    for (int t = 0; t < 9; ++t) {
        float s = 0.f;
        for (int j = 0; j < CC; ++j) s += hid[j] * W2[(c * 9 + t) * CC + j];
        dk[(b * CC + c) * 9 + t] = s;
    }
}

// ---- k1b: weight repack f32[n][ci][3][3] -> bf16 Wp[t][n][ci] --------------
__global__ __launch_bounds__(256) void repack_w(const float* __restrict__ cw,
                                                __bf16* __restrict__ Wp) {
    int i = blockIdx.x * 256 + threadIdx.x;          // i = t*4096 + n*64 + ci
    if (i >= 9 * CC * CC) return;
    int t = i >> 12, rem = i & 4095, n = rem >> 6, ci = rem & 63;
    Wp[i] = (__bf16)cw[(n * CC + ci) * 9 + t];
}

// ---- k2: depthwise dynamic conv + leaky -> bf16 ----------------------------
__global__ __launch_bounds__(128) void dw_conv(const float* __restrict__ x0,
                                               const float* __restrict__ dk,
                                               __bf16* __restrict__ tmp) {
    int idx = blockIdx.x;
    int y = idx & (HH - 1);
    int bc = idx >> 7;                                // uniform -> scalar loads
    float k[9];
#pragma unroll
    for (int t = 0; t < 9; ++t) k[t] = dk[bc * 9 + t];
    int x = threadIdx.x;
    const float* in = x0 + bc * HH * WW;
    float acc = 0.f;
#pragma unroll
    for (int dy = -1; dy <= 1; ++dy) {
        int yy = y + dy;
        if (yy < 0 || yy >= HH) continue;
        const float* row = in + yy * WW;
#pragma unroll
        for (int dx = -1; dx <= 1; ++dx) {
            int xx = x + dx;
            if (xx >= 0 && xx < WW) acc += k[(dy + 1) * 3 + (dx + 1)] * row[xx];
        }
    }
    tmp[bc * HH * WW + y * WW + x] = (__bf16)leakyf(acc);
}

// ---- k3: dense 3x3 conv as implicit GEMM on WMMA bf16 ----------------------
// Grid: B*H*(W/32) blocks x 256 threads (8 waves).
// WG tile: 32 pixels x 64 out-channels of one output row.
// Wave tile: 16 pixels x 16 channels (f32 accum).  K = 64ci x 9 taps = 576.
// All three input rows (y-1, y, y+1) staged to LDS once; single barrier pair;
// 18 WMMAs fully unrolled afterwards.
#define AROW 72           // LDS x-row stride in bf16 (144B: 16B-aligned b128)
#define ARPL (34 * AROW)  // one staged input row plane (xi 0..33)
__global__ __launch_bounds__(256) void dense_conv_wmma(
    const __bf16* __restrict__ tmp, const __bf16* __restrict__ Wp,
    const float* __restrict__ x0, const float* __restrict__ bias,
    float* __restrict__ out) {
    __shared__ __attribute__((aligned(16))) __bf16 As[3 * ARPL]; // [row][xi][ci]

    int idx = blockIdx.x;
    int xt = idx & 3;                 // 32-pixel tile in row
    int y = (idx >> 2) & (HH - 1);
    int b = idx >> 9;
    int xbase = xt * 32;

    int tid = threadIdx.x;
    int wave = tid >> 5;
    int lane = tid & 31;
    int pt = wave >> 2;               // pixel sub-tile (0/1)
    int ct = wave & 3;                // channel sub-tile (0..3)
    int lh = lane >> 4;               // lane half
    int ll = lane & 15;
    int n = ct * 16 + ll;             // this lane's output channel (N)

    const int tmpb = b * CC * HH * WW;

    // Stage rows y-1, y, y+1: ci 0..63, x in [xbase-1, xbase+32]  (zero-pad)
    for (int i = tid; i < 3 * CC * 34; i += 256) {
        int r = i / (CC * 34);
        int rem = i - r * (CC * 34);
        int ci = rem / 34;
        int xi = rem - ci * 34;
        int yy = y + r - 1;
        int gx = xbase - 1 + xi;
        __bf16 v = (__bf16)0.f;
        if (yy >= 0 && yy < HH && gx >= 0 && gx < WW)
            v = tmp[tmpb + (ci * HH + yy) * WW + gx];
        As[r * ARPL + xi * AROW + ci] = v;
    }
    __syncthreads();

    v8f acc = {};
    const __bf16* wrow = Wp + n * CC; // + t*4096 per tap

#pragma unroll
    for (int t = 0; t < 9; ++t) {
        int dy = t / 3 - 1, dx = t % 3 - 1;
        const __bf16* arow = &As[(dy + 1) * ARPL + (pt * 16 + ll + dx + 1) * AROW];
        const __bf16* wtap = wrow + t * 4096;
#pragma unroll
        for (int half = 0; half < 2; ++half) {   // K chunks of 32 within tap
            int cb = half * 32;
            // A 16x32 bf16 layout: elem e, lane -> K=(e&7)+8*lh+16*(e>=8)
            v8bf alo = *(const v8bf*)&arow[cb + 8 * lh];
            v8bf ahi = *(const v8bf*)&arow[cb + 16 + 8 * lh];
            // B 32x16 bf16 layout: lane=N, elem e -> K=(e&7)+8*lh+16*(e>=8)
            v8bf blo = *(const v8bf*)(wtap + cb + 8 * lh);
            v8bf bhi = *(const v8bf*)(wtap + cb + 16 + 8 * lh);
            v16bf a, bm;
#pragma unroll
            for (int e = 0; e < 8; ++e) {
                a[e] = alo[e];  a[e + 8] = ahi[e];
                bm[e] = blo[e]; bm[e + 8] = bhi[e];
            }
            acc = __builtin_amdgcn_wmma_f32_16x16x32_bf16(
                false, a, false, bm, (short)0, acc, false, false);
        }
    }

    // Epilogue: D layout VGPR v <-> M = v + 8*lh  =>  8 consecutive x per lane
    int px = xbase + pt * 16 + lh * 8;
    int obase = ((b * CC + n) * HH + y) * WW + px;
    v4f r0 = *(const v4f*)(x0 + obase);
    v4f r1 = *(const v4f*)(x0 + obase + 4);
    float bz = bias[n];
    v4f o0, o1;
#pragma unroll
    for (int e = 0; e < 4; ++e) {
        o0[e] = acc[e] + bz + r0[e];
        o1[e] = acc[e + 4] + bz + r1[e];
    }
    *(v4f*)(out + obase) = o0;
    *(v4f*)(out + obase + 4) = o1;
}

// ---------------------------------------------------------------------------
extern "C" void kernel_launch(void* const* d_in, const int* in_sizes, int n_in,
                              void* d_out, int out_size, void* d_ws, size_t ws_size,
                              hipStream_t stream) {
    const float* x0 = (const float*)d_in[0];
    const float* d  = (const float*)d_in[1];
    const float* W1 = (const float*)d_in[2];
    const float* W2 = (const float*)d_in[3];
    const float* cw = (const float*)d_in[4];
    const float* cb = (const float*)d_in[5];
    float* out = (float*)d_out;

    char* ws = (char*)d_ws;
    float*  dk  = (float*)ws;
    __bf16* Wp  = (__bf16*)(ws + WS_WP_OFF);
    __bf16* tmp = (__bf16*)(ws + WS_TMP_OFF);

    prep_dyn<<<BB, CC, 0, stream>>>(d, W1, W2, dk);
    repack_w<<<(9 * CC * CC + 255) / 256, 256, 0, stream>>>(cw, Wp);
    dw_conv<<<BB * CC * HH, 128, 0, stream>>>(x0, dk, tmp);
    dense_conv_wmma<<<BB * HH * (WW / 32), 256, 0, stream>>>(tmp, Wp, x0, cb, out);
}
